// FactorizedEnsembleModel_46823733461212
// MI455X (gfx1250) — matrix operand order repaired and checked by reading, not compiled, not run on hardware
//
#include <hip/hip_runtime.h>

typedef __attribute__((ext_vector_type(16))) _Float16 v16h;
typedef __attribute__((ext_vector_type(8)))  _Float16 v8h;
typedef __attribute__((ext_vector_type(8)))  float    v8f;

#define BATCH   4096
#define IN_DIM  40
#define KPAD    64
#define HID     128
#define DE_CNT  330
#define BTILE   128

__device__ __forceinline__ float silu_f(float v) {
    return v / (1.0f + __expf(-v));
}

__device__ __forceinline__ float softplus_f(float v) {
    return (v > 20.0f) ? v : log1pf(__expf(v));
}

// A-matrix fragment (16x32 f16): lane half selects K group of 8; elems 0..7 -> k0..k0+7,
// elems 8..15 -> k0+16..k0+23  (per CDNA5 ISA 16-bit A layout)
__device__ __forceinline__ v16h load_a_frag(const _Float16* base, int stride, int row,
                                            int kbase, int lane) {
    const _Float16* p = base + row * stride + kbase + ((lane >> 4) << 3);
    v8h lo = *(const v8h*)p;
    v8h hi = *(const v8h*)(p + 16);
    v16h f;
#pragma unroll
    for (int i = 0; i < 8; ++i) { f[i] = lo[i]; f[i + 8] = hi[i]; }
    return f;
}

// B-matrix fragment (32x16 f16): lane = column N, lane half selects K group of 8;
// elems 0..7 -> k0..k0+7, elems 8..15 -> k0+16..k0+23 (same K order as A)
__device__ __forceinline__ v16h load_b_frag(const _Float16* base, int stride, int col,
                                            int kbase, int lane) {
    const _Float16* p = base + col * stride + kbase + ((lane >> 4) << 3);
    v8h lo = *(const v8h*)p;
    v8h hi = *(const v8h*)(p + 16);
    v16h f;
#pragma unroll
    for (int i = 0; i < 8; ++i) { f[i] = lo[i]; f[i + 8] = hi[i]; }
    return f;
}

__global__ __launch_bounds__(256)
void ensemble_mlp_wmma_kernel(const float* __restrict__ x,
                              const float* __restrict__ W1, const float* __restrict__ b1,
                              const float* __restrict__ W2, const float* __restrict__ b2,
                              const float* __restrict__ W3, const float* __restrict__ b3,
                              float* __restrict__ out)
{
    // phase A: sOp[0..8191] = X tile (128x64 f16), sOp[8192..16383] = W1 (128x64 f16)
    // phase B: sOp = W2 (128x128 f16)
    __shared__ __align__(16) _Float16 sOp[HID * HID];
    __shared__ __align__(16) _Float16 sH1[BTILE * HID];

    const int tid  = threadIdx.x;
    const int wave = tid >> 5;
    const int lane = tid & 31;
    const int b0   = blockIdx.x * BTILE;
    const int de   = blockIdx.y;

    const float* w1p = W1 + (size_t)de * HID * IN_DIM;
    const float* b1p = b1 + (size_t)de * HID;
    const float* w2p = W2 + (size_t)de * HID * HID;
    const float* b2p = b2 + (size_t)de * HID;
    const float* w3p = W3 + (size_t)de * 2 * HID;
    const float* b3p = b3 + (size_t)de * 2;

    _Float16* sX  = sOp;
    _Float16* sW1 = sOp + BTILE * KPAD;

    // ---- stage X tile and W1 into LDS as f16, K padded 40 -> 64 with zeros
    for (int i = tid; i < BTILE * KPAD; i += 256) {
        int r = i >> 6, k = i & 63;
        float xv = (k < IN_DIM) ? x[(b0 + r) * IN_DIM + k] : 0.0f;
        float wv = (k < IN_DIM) ? w1p[r * IN_DIM + k] : 0.0f;
        sX[i]  = (_Float16)xv;
        sW1[i] = (_Float16)wv;
    }
    __syncthreads();

    const int ncol  = lane & 15;                      // N within 16-wide tile (C/D layout)
    const int arow  = wave * 16 + ncol;               // A row index for fragment loads
    const int rbase = wave * 16 + ((lane >> 4) << 3); // M base for C/D writeback

    v8f acc[8];

    // ================= layer 1: H1 = silu(X * W1^T + b1)  [128 x 128] =================
#pragma unroll
    for (int nt = 0; nt < 8; ++nt) {
        float bv = b1p[nt * 16 + ncol];
#pragma unroll
        for (int i = 0; i < 8; ++i) acc[nt][i] = bv;  // bias folded into accumulator
    }
#pragma unroll
    for (int ks = 0; ks < 2; ++ks) {
        v16h a = load_a_frag(sX, KPAD, arow, ks * 32, lane);
#pragma unroll
        for (int nt = 0; nt < 8; ++nt) {
            v16h bb = load_b_frag(sW1, KPAD, nt * 16 + ncol, ks * 32, lane);
            acc[nt] = __builtin_amdgcn_wmma_f32_16x16x32_f16(
                false, a, false, bb, (short)0, acc[nt], false, false);
        }
    }
    // silu + store H1 (f16) to LDS
#pragma unroll
    for (int nt = 0; nt < 8; ++nt) {
        int col = nt * 16 + ncol;
#pragma unroll
        for (int r = 0; r < 8; ++r) {
            sH1[(rbase + r) * HID + col] = (_Float16)silu_f(acc[nt][r]);
        }
    }
    __syncthreads();

    // ---- stage W2 into LDS (overwrites X/W1 buffer)
    for (int i = tid; i < HID * HID; i += 256)
        sOp[i] = (_Float16)w2p[i];
    __syncthreads();

    // ================= layer 2: H2 = silu(H1 * W2^T + b2)  [128 x 128] =================
#pragma unroll
    for (int nt = 0; nt < 8; ++nt) {
        float bv = b2p[nt * 16 + ncol];
#pragma unroll
        for (int i = 0; i < 8; ++i) acc[nt][i] = bv;
    }
#pragma unroll
    for (int ks = 0; ks < 4; ++ks) {
        v16h a = load_a_frag(sH1, HID, arow, ks * 32, lane);
#pragma unroll
        for (int nt = 0; nt < 8; ++nt) {
            v16h bb = load_b_frag(sOp, HID, nt * 16 + ncol, ks * 32, lane);
            acc[nt] = __builtin_amdgcn_wmma_f32_16x16x32_f16(
                false, a, false, bb, (short)0, acc[nt], false, false);
        }
    }
    // silu in registers (H2 stays fp32)
#pragma unroll
    for (int nt = 0; nt < 8; ++nt)
#pragma unroll
        for (int r = 0; r < 8; ++r)
            acc[nt][r] = silu_f(acc[nt][r]);

    // ================= layer 3: out = H2 * W3^T + b3, then logvar soft-clamp ============
    float w30[8], w31[8];
#pragma unroll
    for (int nt = 0; nt < 8; ++nt) {
        w30[nt] = w3p[nt * 16 + ncol];        // W3[0][h]
        w31[nt] = w3p[HID + nt * 16 + ncol];  // W3[1][h]
    }
    const float b30 = b3p[0], b31 = b3p[1];

#pragma unroll
    for (int r = 0; r < 8; ++r) {
        float p0 = 0.0f, p1 = 0.0f;
#pragma unroll
        for (int nt = 0; nt < 8; ++nt) {
            float h = acc[nt][r];
            p0 += h * w30[nt];
            p1 += h * w31[nt];
        }
        // reduce across the 16-lane half (xor of 1,2,4,8 stays within each half)
#pragma unroll
        for (int off = 1; off < 16; off <<= 1) {
            p0 += __shfl_xor(p0, off, 32);
            p1 += __shfl_xor(p1, off, 32);
        }
        if ((lane & 15) == 0) {
            int row = b0 + rbase + r;  // lanes 0/16 cover M=r and M=8+r
            float mean = p0 + b30;
            float lv   = p1 + b31;
            lv = 5.0f  - softplus_f(5.0f - lv);
            lv = -10.0f + softplus_f(lv + 10.0f);
            out[(size_t)row * DE_CNT + de] = mean;
            out[(size_t)BATCH * DE_CNT + (size_t)row * DE_CNT + de] = lv;
        }
    }
}

extern "C" void kernel_launch(void* const* d_in, const int* in_sizes, int n_in,
                              void* d_out, int out_size, void* d_ws, size_t ws_size,
                              hipStream_t stream) {
    const float* x  = (const float*)d_in[0];
    const float* W1 = (const float*)d_in[1];
    const float* b1 = (const float*)d_in[2];
    const float* W2 = (const float*)d_in[3];
    const float* b2 = (const float*)d_in[4];
    const float* W3 = (const float*)d_in[5];
    const float* b3 = (const float*)d_in[6];

    dim3 grid(BATCH / BTILE, DE_CNT);  // 32 x 330 workgroups
    ensemble_mlp_wmma_kernel<<<grid, 256, 0, stream>>>(
        x, W1, b1, W2, b2, W3, b3, (float*)d_out);
}
